// AtomAttentionEncoder_80917183857336
// MI455X (gfx1250) — matrix operand order repaired and checked by reading, not compiled, not run on hardware
//
#include <hip/hip_runtime.h>

#define N_ATOMS 1536
#define C_ATOM  128
#define C_PAIRD 16
#define N_TOKS  384
#define C_TOKD  384
#define C_HIDD  32
#define NHEAD   4
#define NLAY    3
#define NQW     32
#define NKW     128
#define NBLK    48      /* N_ATOMS / NQW */
#define INFV    1e9f

typedef __attribute__((ext_vector_type(16))) _Float16 v16h;
typedef __attribute__((ext_vector_type(8)))  float    v8f;
typedef __attribute__((ext_vector_type(4)))  float    v4f;

__device__ inline float sigmoid_(float x){ return 1.f/(1.f+__expf(-x)); }

__device__ inline v16h cvt16(v4f a0, v4f a1, v4f a2, v4f a3)
{
  v16h r;
  #pragma unroll
  for (int j=0;j<4;++j){
    r[j]    = (_Float16)a0[j];
    r[4+j]  = (_Float16)a1[j];
    r[8+j]  = (_Float16)a2[j];
    r[12+j] = (_Float16)a3[j];
  }
  return r;
}

enum GemmMode { M_PLAIN=0, M_SIGMOID=1, M_RELU=2, M_ADALN=3, M_SILUMUL=4, M_EGATE=5 };

// Load one 16x32 f16 A fragment (ISA 7.12.2 16-bit A layout) with vector loads.
//   e 0..7  -> K = k0 + 8*half + e        (contiguous 8 floats)
//   e 8..15 -> K = k0 + 16 + 8*half + e-8 (contiguous 8 floats)
template<bool HS>
__device__ inline v16h load_afrag(const float* __restrict__ A, int row, int K,
                                  int k0, int half, const float* __restrict__ sc)
{
  const float* p = A + row*K + k0 + half*8;
  v4f a0 = *(const v4f*)(p);
  v4f a1 = *(const v4f*)(p + 4);
  v4f a2 = *(const v4f*)(p + 16);
  v4f a3 = *(const v4f*)(p + 20);
  if (HS){
    const float* s = sc + k0 + half*8;
    a0 *= *(const v4f*)(s);
    a1 *= *(const v4f*)(s + 4);
    a2 *= *(const v4f*)(s + 16);
    a3 *= *(const v4f*)(s + 20);
  }
  return cvt16(a0,a1,a2,a3);
}

// ---------------------------------------------------------------------------
// Generic fused GEMM: 32x16 output tile per wave (2 A frags share each B frag),
// K stepped by 32 via v_wmma_f32_16x16x32_f16. Optional second weight matrix
// (DUAL) reuses both A frags -> 4 wmma per k-step.
// ---------------------------------------------------------------------------
template<int MODE, bool ACC, bool DUAL, bool HS>
__global__ __launch_bounds__(128)
void gemm_k(const float* __restrict__ A, const float* __restrict__ ascale,
            const float* __restrict__ W1, const float* __restrict__ b1,
            const float* __restrict__ W2, const float* __restrict__ E,
            float* __restrict__ out, int Mdim, int K, int N)
{
  const int wave = threadIdx.x >> 5;
  const int lane = threadIdx.x & 31;
  const int half = lane >> 4;
  const int l16  = lane & 15;
  const int tilesN = N >> 4;
  const int tilesTotal = (Mdim >> 5) * tilesN;   // 32-row x 16-col tiles
  const int tile = blockIdx.x * 4 + wave;
  if (tile >= tilesTotal) return;
  const int tM = tile / tilesN, tN = tile % tilesN;
  const int row0 = tM*32 + l16;
  const int row1 = row0 + 16;
  const int colB = tN*16 + l16;

  v8f a10, a11, a20, a21;
  #pragma unroll
  for (int r=0;r<8;++r){ a10[r]=0.f; a11[r]=0.f; a20[r]=0.f; a21[r]=0.f; }

  for (int k0=0; k0<K; k0+=32){
    v16h af0 = load_afrag<HS>(A, row0, K, k0, half, ascale);
    v16h af1 = load_afrag<HS>(A, row1, K, k0, half, ascale);
    v16h bf1, bf2;
    #pragma unroll
    for (int e=0;e<16;++e){
      int kk = k0 + half*8 + e + (e & 8);
      bf1[e] = (_Float16)W1[kk*N + colB];
      if (DUAL) bf2[e] = (_Float16)W2[kk*N + colB];
    }
    a10 = __builtin_amdgcn_wmma_f32_16x16x32_f16(false, af0, false, bf1, (short)0, a10, false, false);
    a11 = __builtin_amdgcn_wmma_f32_16x16x32_f16(false, af1, false, bf1, (short)0, a11, false, false);
    if (DUAL){
      a20 = __builtin_amdgcn_wmma_f32_16x16x32_f16(false, af0, false, bf2, (short)0, a20, false, false);
      a21 = __builtin_amdgcn_wmma_f32_16x16x32_f16(false, af1, false, bf2, (short)0, a21, false, false);
    }
  }

  const float bias = (MODE==M_PLAIN || MODE==M_SIGMOID)
                       ? (b1 ? b1[colB] : 0.f)
                       : (MODE==M_ADALN ? b1[colB] : 0.f);
  #pragma unroll
  for (int rb=0; rb<2; ++rb){
    #pragma unroll
    for (int r=0;r<8;++r){
      int row = tM*32 + rb*16 + r + 8*half;
      float x = rb ? a11[r] : a10[r];
      float y = rb ? a21[r] : a20[r];
      float res;
      if constexpr (MODE == M_PLAIN)        res = x + bias;
      else if constexpr (MODE == M_SIGMOID) res = sigmoid_(x + bias);
      else if constexpr (MODE == M_RELU)    res = fmaxf(x, 0.f);
      else if constexpr (MODE == M_ADALN)   res = sigmoid_(x + bias) * E[row*N+colB] + y;
      else if constexpr (MODE == M_SILUMUL) res = (x * sigmoid_(x)) * y;
      else                                  res = E[row*N+colB] * x;   // EGATE
      if constexpr (ACC) out[row*N+colB] += res;
      else               out[row*N+colB]  = res;
    }
  }
}

// ---------------------------------------------------------------------------
// Feature embed: cl[n,c] = feats[n,:390] @ W_feats (structured pieces)
// ---------------------------------------------------------------------------
__global__ __launch_bounds__(128)
void embed_k(const float* __restrict__ pos, const float* __restrict__ msk,
             const float* __restrict__ elem, const float* __restrict__ chg,
             const float* __restrict__ chars, const float* __restrict__ uid,
             const float* __restrict__ Wf, float* __restrict__ cl)
{
  int n = blockIdx.x, c = threadIdx.x;
  float acc = pos[n*3+0]*Wf[0*C_ATOM+c] + pos[n*3+1]*Wf[1*C_ATOM+c]
            + pos[n*3+2]*Wf[2*C_ATOM+c] + msk[n]*Wf[3*C_ATOM+c];
  for (int e=0;e<128;++e) acc += elem[n*128+e]*Wf[(4+e)*C_ATOM+c];
  acc += chg[n]*Wf[132*C_ATOM+c];
  for (int j=0;j<256;++j) acc += chars[n*256+j]*Wf[(133+j)*C_ATOM+c];
  acc += uid[n]*Wf[389*C_ATOM+c];
  cl[n*C_ATOM+c] = acc;
}

// LayerNorm over C_ATOM (no affine)
__global__ __launch_bounds__(128)
void ln_k(const float* __restrict__ src, float* __restrict__ dst)
{
  __shared__ float red[128];
  int n = blockIdx.x, t = threadIdx.x;
  float x = src[n*C_ATOM+t];
  red[t] = x; __syncthreads();
  #pragma unroll
  for (int s=64; s>0; s>>=1){ if (t<s) red[t]+=red[t+s]; __syncthreads(); }
  float mu = red[0] * (1.f/128.f);
  __syncthreads();
  float d = x - mu;
  red[t] = d*d; __syncthreads();
  #pragma unroll
  for (int s=64; s>0; s>>=1){ if (t<s) red[t]+=red[t+s]; __syncthreads(); }
  float var = red[0] * (1.f/128.f);
  dst[n*C_ATOM+t] = d * rsqrtf(var + 1e-5f);
}

// cwl/cwm: relu(cl) @ W_l / W_m  -> [N,16] each
__global__ __launch_bounds__(32)
void pairprep_k(const float* __restrict__ cl, const float* __restrict__ Wl,
                const float* __restrict__ Wm, float* __restrict__ cwl,
                float* __restrict__ cwm)
{
  int n = blockIdx.x, t = threadIdx.x;
  const float* W = (t < 16) ? Wl : Wm;
  int c = t & 15;
  float acc = 0.f;
  for (int j=0;j<C_ATOM;++j) acc += fmaxf(cl[n*C_ATOM+j],0.f) * W[j*16+c];
  if (t < 16) cwl[n*16+c] = acc; else cwm[n*16+c] = acc;
}

__global__ void amask_k(const float* __restrict__ a2t, const float* __restrict__ tmask,
                        float* __restrict__ am)
{
  int n = blockIdx.x*256 + threadIdx.x;
  if (n >= N_ATOMS) return;
  float s = 0.f;
  for (int t=0;t<N_TOKS;++t) s += a2t[n*N_TOKS+t]*tmask[t];
  am[n] = s;
}

// ---------------------------------------------------------------------------
// Block-sparse pair pipeline fused to attention biases (beta/mask folded)
// ---------------------------------------------------------------------------
__global__ __launch_bounds__(128)
void pairbias_k(const float* __restrict__ pos, const float* __restrict__ uid,
                const float* __restrict__ cwl, const float* __restrict__ cwm,
                const float* __restrict__ am,
                const float* __restrict__ Wro, const float* __restrict__ Winv,
                const float* __restrict__ Wval,
                const float* __restrict__ W1, const float* __restrict__ W2,
                const float* __restrict__ W3,
                const float* __restrict__ lnzg, const float* __restrict__ lnzb,
                const float* __restrict__ Wb, float* __restrict__ zb)
{
  __shared__ float sW1[256], sW2[256], sW3[256];
  __shared__ float sWro[48], sWinv[16], sWval[16];
  __shared__ float sg[48], sbv[48], sWb[192];
  __shared__ float Pl[NQW][4];
  __shared__ float Cl[NQW][16];
  int b = blockIdx.x, t = threadIdx.x;
  for (int i=t;i<256;i+=128){ sW1[i]=W1[i]; sW2[i]=W2[i]; sW3[i]=W3[i]; }
  if (t<48){ sWro[t]=Wro[t]; sg[t]=lnzg[t]; sbv[t]=lnzb[t]; }
  if (t<16){ sWinv[t]=Winv[t]; sWval[t]=Wval[t]; }
  for (int i=t;i<192;i+=128) sWb[i]=Wb[i];
  if (t<NQW){
    int l = b*NQW + t;
    Pl[t][0]=pos[l*3+0]; Pl[t][1]=pos[l*3+1]; Pl[t][2]=pos[l*3+2]; Pl[t][3]=uid[l];
  }
  for (int i=t;i<NQW*16;i+=128) Cl[i>>4][i&15] = cwl[(b*NQW+(i>>4))*16 + (i&15)];
  __syncthreads();

  int kk = t;                       // key slot 0..127
  int m  = b*NQW - 48 + kk;
  bool valid = (m >= 0 && m < N_ATOMS);
  float pmx=0,pmy=0,pmz=0,um=0,amk=1.f,cm[16];
  #pragma unroll
  for (int c=0;c<16;++c) cm[c]=0.f;
  if (valid){
    pmx=pos[m*3+0]; pmy=pos[m*3+1]; pmz=pos[m*3+2]; um=uid[m]; amk=am[m];
    #pragma unroll
    for (int c=0;c<16;++c) cm[c]=cwm[m*16+c];
  }
  for (int qi=0; qi<NQW; ++qi){
    if (!valid){
      #pragma unroll
      for (int i=0;i<NLAY;++i)
        #pragma unroll
        for (int h=0;h<NHEAD;++h)
          zb[(((i*NBLK+b)*NHEAD+h)*NQW + qi)*NKW + kk] = -INFV;
      continue;
    }
    float dx=pmx-Pl[qi][0], dy=pmy-Pl[qi][1], dz=pmz-Pl[qi][2];
    float vf = (um == Pl[qi][3]) ? 1.f : 0.f;
    float isq = 1.f/(1.f + dx*dx+dy*dy+dz*dz);
    float p[16], t1[16], t2[16];
    #pragma unroll
    for (int c=0;c<16;++c)
      p[c] = (dx*sWro[c] + dy*sWro[16+c] + dz*sWro[32+c] + isq*sWinv[c] + sWval[c])*vf
           + Cl[qi][c] + cm[c];
    #pragma unroll
    for (int c2=0;c2<16;++c2){ float a=0;
      #pragma unroll
      for (int c=0;c<16;++c) a += fmaxf(p[c],0.f)*sW1[c*16+c2];
      t1[c2]=a; }
    #pragma unroll
    for (int c2=0;c2<16;++c2){ float a=0;
      #pragma unroll
      for (int c=0;c<16;++c) a += fmaxf(t1[c],0.f)*sW2[c*16+c2];
      t2[c2]=a; }
    #pragma unroll
    for (int c2=0;c2<16;++c2){ float a=0;
      #pragma unroll
      for (int c=0;c<16;++c) a += fmaxf(t2[c],0.f)*sW3[c*16+c2];
      p[c2]+=a; }
    float mu=0;
    #pragma unroll
    for (int c=0;c<16;++c) mu+=p[c];
    mu *= (1.f/16.f);
    float var=0;
    #pragma unroll
    for (int c=0;c<16;++c){ float d=p[c]-mu; var+=d*d; }
    float rs = rsqrtf(var*(1.f/16.f) + 1e-5f);
    float maskterm = (amk - 1.f) * INFV;
    #pragma unroll
    for (int i=0;i<NLAY;++i){
      float yy[16];
      #pragma unroll
      for (int c=0;c<16;++c) yy[c] = (p[c]-mu)*rs*sg[i*16+c] + sbv[i*16+c];
      #pragma unroll
      for (int h=0;h<NHEAD;++h){
        float z=0;
        #pragma unroll
        for (int c=0;c<16;++c) z += yy[c]*sWb[(i*16+c)*NHEAD+h];
        zb[(((i*NBLK+b)*NHEAD+h)*NQW + qi)*NKW + kk] = z + maskterm;
      }
    }
  }
}

// ---------------------------------------------------------------------------
// Block-local WMMA attention: one WG (4 waves) per (query block, head).
// Phase A: S = (Q*s) @ K^T + zb  -- Q/K fragments loaded straight from global
//          (each wave: 2 Q-frags x 2 K-frags -> 4 wmma, K=32 single step)
// softmax rows in LDS, then
// Phase B: O = P @ V per 16x16 tile (4 k-steps -> 4 wmma per wave),
//          gate-multiplied on store.
// ---------------------------------------------------------------------------
__global__ __launch_bounds__(128)
void attn_k(const float* __restrict__ q, const float* __restrict__ k,
            const float* __restrict__ v, const float* __restrict__ gt,
            const float* __restrict__ zbL, float* __restrict__ og)
{
  __shared__ float Vs[NKW][33];
  __shared__ float S[NQW][132];
  const int b = blockIdx.x, h = blockIdx.y, t = threadIdx.x;
  const int w = t >> 5, lane = t & 31;
  const int half = lane >> 4, l16 = lane & 15;
  const int ks0 = b*NQW - 48;
  const float* zp = zbL + (((size_t)b*NHEAD + h)*NQW)*NKW;

  // stage V (zero-filled outside the window)
  for (int idx=t; idx<NKW*32; idx+=128){
    int kk=idx>>5, c=idx&31;
    int m = ks0 + kk;
    float vv = 0.f;
    if (m>=0 && m<N_ATOMS) vv = v[m*C_ATOM+h*32+c];
    Vs[kk][c] = vv;
  }

  // ---- Phase A: logits ----
  v16h qf[2];
  #pragma unroll
  for (int mt=0; mt<2; ++mt){
    int atom = b*NQW + mt*16 + l16;
    const float* p = q + atom*C_ATOM + h*32 + half*16;
    v4f q0 = *(const v4f*)(p)      * 0.1767766953f;
    v4f q1 = *(const v4f*)(p + 4)  * 0.1767766953f;
    v4f q2 = *(const v4f*)(p + 8)  * 0.1767766953f;
    v4f q3 = *(const v4f*)(p + 12) * 0.1767766953f;
    qf[mt] = cvt16(q0,q1,q2,q3);
  }
  #pragma unroll
  for (int nt=0; nt<2; ++nt){
    const int n   = w*2 + nt;            // N-tile 0..7
    const int key = n*16 + l16;
    const int m   = ks0 + key;
    const float sel = (m>=0 && m<N_ATOMS) ? 1.f : 0.f;
    const int mc = m < 0 ? 0 : (m >= N_ATOMS ? N_ATOMS-1 : m);
    const float* p = k + mc*C_ATOM + h*32 + half*16;   // 16 contiguous floats
    v4f k0v = *(const v4f*)(p)      * sel;
    v4f k1v = *(const v4f*)(p + 4)  * sel;
    v4f k2v = *(const v4f*)(p + 8)  * sel;
    v4f k3v = *(const v4f*)(p + 12) * sel;
    v16h kf = cvt16(k0v,k1v,k2v,k3v);
    #pragma unroll
    for (int mt=0; mt<2; ++mt){
      v8f acc;
      #pragma unroll
      for (int r=0;r<8;++r) acc[r]=0.f;
      acc = __builtin_amdgcn_wmma_f32_16x16x32_f16(false, qf[mt], false, kf,
                                                   (short)0, acc, false, false);
      #pragma unroll
      for (int r=0;r<8;++r){
        int row = mt*16 + r + 8*half;
        S[row][key] = acc[r] + zp[row*NKW + key];
      }
    }
  }
  __syncthreads();

  // ---- softmax rows ----
  if (t < NQW){
    float mx=-3.4e38f;
    for (int kk=0;kk<NKW;++kk) mx = fmaxf(mx, S[t][kk]);
    float sum=0.f;
    for (int kk=0;kk<NKW;++kk){ float e=__expf(S[t][kk]-mx); S[t][kk]=e; sum+=e; }
    float inv=1.f/sum;
    for (int kk=0;kk<NKW;++kk) S[t][kk]*=inv;
  }
  __syncthreads();

  // ---- Phase B: O = P @ V (one 16x16 tile per wave) ----
  {
    const int mw = w >> 1, nw = w & 1;
    v8f acc;
    #pragma unroll
    for (int r=0;r<8;++r) acc[r]=0.f;
    const float* sp = &S[mw*16 + l16][0];
    #pragma unroll
    for (int k0=0; k0<NKW; k0+=32){
      v4f p0 = *(const v4f*)(sp + k0 + half*8);
      v4f p1 = *(const v4f*)(sp + k0 + half*8 + 4);
      v4f p2 = *(const v4f*)(sp + k0 + 16 + half*8);
      v4f p3 = *(const v4f*)(sp + k0 + 16 + half*8 + 4);
      v16h pf = cvt16(p0,p1,p2,p3);
      v16h vf;
      #pragma unroll
      for (int e=0;e<16;++e){
        int key = k0 + 8*half + e + (e & 8);
        vf[e] = (_Float16)Vs[key][nw*16 + l16];
      }
      acc = __builtin_amdgcn_wmma_f32_16x16x32_f16(false, pf, false, vf,
                                                   (short)0, acc, false, false);
    }
    #pragma unroll
    for (int r=0;r<8;++r){
      int qi  = mw*16 + r + 8*half;
      int c   = h*32 + nw*16 + l16;
      int row = b*NQW + qi;
      og[row*C_ATOM + c] = gt[row*C_ATOM + c] * acc[r];
    }
  }
}

__global__ void counts_k(const float* __restrict__ a2t, float* __restrict__ cnt)
{
  int t = blockIdx.x*256 + threadIdx.x;
  if (t >= N_TOKS) return;
  float s=0.f;
  for (int n=0;n<N_ATOMS;++n) s += a2t[n*N_TOKS+t];
  cnt[t] = fmaxf(s, 1.f);
}

__global__ __launch_bounds__(128)
void agg_k(const float* __restrict__ a2t, const float* __restrict__ al,
           const float* __restrict__ cnt, float* __restrict__ out)
{
  int tok = blockIdx.x, t = threadIdx.x;
  float a0=0.f,a1=0.f,a2=0.f;
  for (int n=0;n<N_ATOMS;++n){
    float w = a2t[n*N_TOKS+tok];
    if (w != 0.f){
      a0 += w*al[n*C_TOKD+t];
      a1 += w*al[n*C_TOKD+t+128];
      a2 += w*al[n*C_TOKD+t+256];
    }
  }
  float inv = 1.f/cnt[tok];
  out[tok*C_TOKD+t    ] = a0*inv;
  out[tok*C_TOKD+t+128] = a1*inv;
  out[tok*C_TOKD+t+256] = a2*inv;
}

// ---------------------------------------------------------------------------
static void gemm(hipStream_t s, int mode, bool acc,
                 const float* A, const float* ascale,
                 const float* W1, const float* b1,
                 const float* W2, const float* E,
                 float* out, int M, int K, int N)
{
  dim3 g(((M/32)*(N/16) + 3)/4), bl(128);
  if (mode == M_ADALN)
    gemm_k<M_ADALN,false,true ,true ><<<g,bl,0,s>>>(A,ascale,W1,b1,W2,E,out,M,K,N);
  else if (mode == M_SILUMUL)
    gemm_k<M_SILUMUL,false,true ,false><<<g,bl,0,s>>>(A,ascale,W1,b1,W2,E,out,M,K,N);
  else if (mode == M_EGATE && acc)
    gemm_k<M_EGATE,true ,false,false><<<g,bl,0,s>>>(A,ascale,W1,b1,W2,E,out,M,K,N);
  else if (mode == M_EGATE)
    gemm_k<M_EGATE,false,false,false><<<g,bl,0,s>>>(A,ascale,W1,b1,W2,E,out,M,K,N);
  else if (mode == M_SIGMOID)
    gemm_k<M_SIGMOID,false,false,false><<<g,bl,0,s>>>(A,ascale,W1,b1,W2,E,out,M,K,N);
  else if (mode == M_RELU)
    gemm_k<M_RELU,false,false,false><<<g,bl,0,s>>>(A,ascale,W1,b1,W2,E,out,M,K,N);
  else
    gemm_k<M_PLAIN,false,false,false><<<g,bl,0,s>>>(A,ascale,W1,b1,W2,E,out,M,K,N);
}

extern "C" void kernel_launch(void* const* d_in, const int* in_sizes, int n_in,
                              void* d_out, int out_size, void* d_ws, size_t ws_size,
                              hipStream_t stream)
{
  (void)in_sizes; (void)n_in; (void)out_size; (void)ws_size;
  const float* ref_pos   = (const float*)d_in[0];
  const float* ref_mask  = (const float*)d_in[1];
  const float* ref_elem  = (const float*)d_in[2];
  const float* ref_chg   = (const float*)d_in[3];
  const float* ref_chars = (const float*)d_in[4];
  const float* ref_uid   = (const float*)d_in[5];
  const float* token_msk = (const float*)d_in[6];
  const float* a2t       = (const float*)d_in[7];
  const float* W_feats   = (const float*)d_in[8];
  const float* W_ro      = (const float*)d_in[9];
  const float* W_isq     = (const float*)d_in[10];
  const float* W_val     = (const float*)d_in[11];
  const float* W_l       = (const float*)d_in[12];
  const float* W_m       = (const float*)d_in[13];
  const float* W_mlp1    = (const float*)d_in[14];
  const float* W_mlp2    = (const float*)d_in[15];
  const float* W_mlp3    = (const float*)d_in[16];
  const float* W_out_tok = (const float*)d_in[17];
  const float* ada_g     = (const float*)d_in[18];
  const float* ada_Wg    = (const float*)d_in[19];
  const float* ada_bg    = (const float*)d_in[20];
  const float* ada_Ws    = (const float*)d_in[21];
  const float* Wq        = (const float*)d_in[22];
  const float* bq        = (const float*)d_in[23];
  const float* Wk        = (const float*)d_in[24];
  const float* Wv        = (const float*)d_in[25];
  const float* lnz_g     = (const float*)d_in[26];
  const float* lnz_b     = (const float*)d_in[27];
  const float* Wb        = (const float*)d_in[28];
  const float* Wgate     = (const float*)d_in[29];
  const float* Wo        = (const float*)d_in[30];
  const float* Wsg       = (const float*)d_in[31];
  const float* bsg       = (const float*)d_in[32];
  const float* trada_g   = (const float*)d_in[33];
  const float* trada_Wg  = (const float*)d_in[34];
  const float* trada_bg  = (const float*)d_in[35];
  const float* trada_Ws  = (const float*)d_in[36];
  const float* trW1      = (const float*)d_in[37];
  const float* trW2      = (const float*)d_in[38];
  const float* trWog     = (const float*)d_in[39];
  const float* trbog     = (const float*)d_in[40];
  const float* trWout    = (const float*)d_in[41];
  float* out = (float*)d_out;

  // workspace carve (256B aligned)
  size_t off = 0;
  char* base = (char*)d_ws;
  auto carve = [&](size_t nfloat)->float*{
    float* p = (float*)(base + off);
    off = (off + nfloat*sizeof(float) + 255) & ~(size_t)255;
    return p;
  };
  const size_t NC = (size_t)N_ATOMS*C_ATOM;
  float* cl    = carve(NC);
  float* ln_cl = carve(NC);
  float* ln_a  = carve(NC);
  float* cwl   = carve((size_t)N_ATOMS*16);
  float* cwm   = carve((size_t)N_ATOMS*16);
  float* amsk  = carve(N_ATOMS);
  float* zb    = carve((size_t)NLAY*NBLK*NHEAD*NQW*NKW);
  float* an    = carve(NC);
  float* qb    = carve(NC);
  float* kb    = carve(NC);
  float* vb    = carve(NC);
  float* gb    = carve(NC);
  float* sgA   = carve(NC);
  float* sgB   = carve(NC);
  float* og    = carve(NC);
  float* tn    = carve(NC);
  float* hh    = carve((size_t)N_ATOMS*256);
  float* bufA  = carve(NC);
  float* bufB  = carve(NC);
  float* al    = carve((size_t)N_ATOMS*C_TOKD);
  float* cnt   = carve(N_TOKS);

  // 1) embed + LN(cl) + pair conditioning vectors + atom mask
  embed_k<<<N_ATOMS,128,0,stream>>>(ref_pos, ref_mask, ref_elem, ref_chg,
                                    ref_chars, ref_uid, W_feats, cl);
  ln_k<<<N_ATOMS,128,0,stream>>>(cl, ln_cl);
  pairprep_k<<<N_ATOMS,32,0,stream>>>(cl, W_l, W_m, cwl, cwm);
  amask_k<<<(N_ATOMS+255)/256,256,0,stream>>>(a2t, token_msk, amsk);

  // 2) block-sparse pair pipeline -> fused attention biases (beta/mask folded)
  pairbias_k<<<NBLK,128,0,stream>>>(ref_pos, ref_uid, cwl, cwm, amsk,
                                    W_ro, W_isq, W_val, W_mlp1, W_mlp2, W_mlp3,
                                    lnz_g, lnz_b, Wb, zb);

  // 3) diffusion transformer layers
  const float* a_cur = cl;
  float* bufs[2] = { bufA, bufB };
  for (int i=0;i<NLAY;++i){
    float* a_nxt = bufs[i & 1];
    const int W128 = 128*128;
    ln_k<<<N_ATOMS,128,0,stream>>>(a_cur, ln_a);
    // attention adaLN (dual GEMM), Q/K/V/gate projections, sigma gate
    gemm(stream, M_ADALN, false, ln_cl, ada_g+i*128, ada_Wg+i*W128, ada_bg+i*128,
         ada_Ws+i*W128, ln_a, an, N_ATOMS, 128, 128);
    gemm(stream, M_PLAIN,   false, an, nullptr, Wq+i*W128, bq+i*128, nullptr, nullptr, qb, N_ATOMS,128,128);
    gemm(stream, M_PLAIN,   false, an, nullptr, Wk+i*W128, nullptr, nullptr, nullptr, kb, N_ATOMS,128,128);
    gemm(stream, M_PLAIN,   false, an, nullptr, Wv+i*W128, nullptr, nullptr, nullptr, vb, N_ATOMS,128,128);
    gemm(stream, M_SIGMOID, false, an, nullptr, Wgate+i*W128, nullptr, nullptr, nullptr, gb, N_ATOMS,128,128);
    gemm(stream, M_SIGMOID, false, cl, nullptr, Wsg+i*W128, bsg+i*128, nullptr, nullptr, sgA, N_ATOMS,128,128);
    // block-local WMMA attention (gate fused into output)
    attn_k<<<dim3(NBLK,NHEAD),128,0,stream>>>(qb, kb, vb, gb,
        zb + (size_t)i*NBLK*NHEAD*NQW*NKW, og);
    // attn_out = sgA * (og @ Wo)
    gemm(stream, M_EGATE, false, og, nullptr, Wo+i*W128, nullptr, nullptr, sgA,
         a_nxt, N_ATOMS, 128, 128);
    // transition branch (on pre-attention a)
    gemm(stream, M_ADALN, false, ln_cl, trada_g+i*128, trada_Wg+i*W128, trada_bg+i*128,
         trada_Ws+i*W128, ln_a, tn, N_ATOMS, 128, 128);
    gemm(stream, M_SILUMUL, false, tn, nullptr, trW1+i*128*256, nullptr,
         trW2+i*128*256, nullptr, hh, N_ATOMS, 128, 256);
    gemm(stream, M_SIGMOID, false, cl, nullptr, trWog+i*W128, trbog+i*128,
         nullptr, nullptr, sgB, N_ATOMS, 128, 128);
    // a = attn_out + sgB * (hh @ Wout)
    gemm(stream, M_EGATE, true, hh, nullptr, trWout+i*256*128, nullptr, nullptr, sgB,
         a_nxt, N_ATOMS, 256, 128);
    a_cur = a_nxt;
  }

  // 4) atoms -> tokens
  gemm(stream, M_RELU, false, a_cur, nullptr, W_out_tok, nullptr, nullptr, nullptr,
       al, N_ATOMS, 128, C_TOKD);
  counts_k<<<(N_TOKS+255)/256,256,0,stream>>>(a2t, cnt);
  agg_k<<<N_TOKS,128,0,stream>>>(a2t, al, cnt, out);
}
